// Encoder_46901042872556
// MI455X (gfx1250) — matrix operand order repaired and checked by reading, not compiled, not run on hardware
//
#include <hip/hip_runtime.h>

// ---------------------------------------------------------------------------
// Transformer encoder layer for MI455X (gfx1250, wave32, WMMA).
// B=4, S=4096, D=512, F=2048.  All heavy math on v_wmma_f32_16x16x32_bf16.
// Attention keeps the 16x4096 score strip entirely in LDS (320KB/WGP).
// ---------------------------------------------------------------------------

#define B_DIM   4
#define S_DIM   4096
#define DM      512
#define DF      2048
#define M_ROWS  (B_DIM * S_DIM)   // 16384
#define LN_EPS  1e-5f

typedef __bf16 bf16_t;
typedef __attribute__((ext_vector_type(16))) __bf16 v16bf;
typedef __attribute__((ext_vector_type(8)))  __bf16 v8bf;
typedef __attribute__((ext_vector_type(8)))  float  v8f;

__device__ inline v8f v8f_zero() {
    v8f z;
#pragma unroll
    for (int i = 0; i < 8; ++i) z[i] = 0.0f;
    return z;
}

__device__ inline v8f wmma_bf16(v16bf a, v16bf b, v8f c) {
    // D = A(16x32 bf16) x B(32x16 bf16) + C(16x16 f32)
    return __builtin_amdgcn_wmma_f32_16x16x32_bf16(
        /*neg_a=*/false, a, /*neg_b=*/false, b,
        /*c_mod=*/(short)0, c, /*reuse_a=*/false, /*reuse_b=*/false);
}

// A-fragment (16x32, MxK): lane&15 = M row; lane>=16 selects K-phase.
// Per lane: K = {khalf*8 .. +8} and {16+khalf*8 .. +8}  (two 16B runs).
__device__ inline v16bf load_a_frag(const bf16_t* base, int ld, int row, int k0, int khalf) {
    const bf16_t* p = base + (size_t)row * ld + k0 + khalf * 8;
    v8bf lo = *(const v8bf*)(p);
    v8bf hi = *(const v8bf*)(p + 16);
    v16bf r;
#pragma unroll
    for (int i = 0; i < 8; ++i) { r[i] = lo[i]; r[i + 8] = hi[i]; }
    return r;
}

// A-fragment sourced from f32 (converted to bf16 on the fly).
__device__ inline v16bf load_a_frag(const float* base, int ld, int row, int k0, int khalf) {
    const float* p = base + (size_t)row * ld + k0 + khalf * 8;
    v8f lo = *(const v8f*)(p);
    v8f hi = *(const v8f*)(p + 16);
    v16bf r;
#pragma unroll
    for (int i = 0; i < 8; ++i) { r[i] = (bf16_t)lo[i]; r[i + 8] = (bf16_t)hi[i]; }
    return r;
}

// B-fragment (32x16, KxN): lane&15 = N column; per lane K = {khalf*8 .. +8} and {16+khalf*8 .. +8}
// (two 16B runs).  `base` is the KxN matrix stored transposed [N][K].
__device__ inline v16bf load_b_frag(const bf16_t* baseT, int ld, int nrow, int k0, int khalf) {
    const bf16_t* p = baseT + (size_t)nrow * ld + k0 + khalf * 8;
    v8bf lo = *(const v8bf*)(p);
    v8bf hi = *(const v8bf*)(p + 16);
    v16bf r;
#pragma unroll
    for (int i = 0; i < 8; ++i) { r[i] = lo[i]; r[i + 8] = hi[i]; }
    return r;
}

// ---------------------------------------------------------------------------
// Weight prep: W[K][N] f32  ->  Wt[N][K] bf16   (tiled transpose via LDS)
// ---------------------------------------------------------------------------
__global__ __launch_bounds__(256)
void transpose_w_kernel(const float* __restrict__ W, bf16_t* __restrict__ Wt,
                        int K, int N) {
    __shared__ float tile[32][33];
    int k0 = blockIdx.x * 32, n0 = blockIdx.y * 32;
    int tx = threadIdx.x & 31, ty = threadIdx.x >> 5;   // 32x8
#pragma unroll
    for (int i = ty; i < 32; i += 8)
        tile[i][tx] = W[(size_t)(k0 + i) * N + (n0 + tx)];
    __syncthreads();
#pragma unroll
    for (int i = ty; i < 32; i += 8)
        Wt[(size_t)(n0 + i) * K + (k0 + tx)] = (bf16_t)tile[tx][i];
}

// V[b][s][d] bf16 -> Vt[b][d][s] bf16
__global__ __launch_bounds__(256)
void transpose_v_kernel(const bf16_t* __restrict__ Vb, bf16_t* __restrict__ Vt) {
    __shared__ bf16_t tile[32][33];
    int b  = blockIdx.z;
    int s0 = blockIdx.x * 32, d0 = blockIdx.y * 32;
    int tx = threadIdx.x & 31, ty = threadIdx.x >> 5;
    const bf16_t* src = Vb + (size_t)b * S_DIM * DM;
    bf16_t*       dst = Vt + (size_t)b * DM * S_DIM;
#pragma unroll
    for (int i = ty; i < 32; i += 8)
        tile[i][tx] = src[(size_t)(s0 + i) * DM + (d0 + tx)];
    __syncthreads();
#pragma unroll
    for (int i = ty; i < 32; i += 8)
        dst[(size_t)(d0 + i) * S_DIM + (s0 + tx)] = tile[tx][i];
}

// ---------------------------------------------------------------------------
// GEMM:  C[M][N] = act( A[M][K] @ W[K][N] + bias )
//   A: f32 or bf16 row-major; W given as Wt[N][K] bf16.
//   Block: 256 thr = 8 waves; block tile 16 x 512; wave tile 16 x 64
//   (one A fragment reused across 4 B fragments -> 4 accumulators).
// ---------------------------------------------------------------------------
template <typename AT, bool RELU, bool OUT_BF16>
__global__ __launch_bounds__(256)
void gemm_bias_kernel(const AT* __restrict__ A, const bf16_t* __restrict__ Wt,
                      const float* __restrict__ bias, void* __restrict__ out,
                      int M, int N, int K) {
    const int tid   = threadIdx.x;
    const int wave  = tid >> 5;
    const int lane  = tid & 31;
    const int lrow  = lane & 15;
    const int khalf = lane >> 4;

    const int nBlkN = N >> 9;                    // N / 512
    const int mt = blockIdx.x / nBlkN;
    const int nb = blockIdx.x % nBlkN;
    const int m0 = mt * 16;
    const int c0 = nb * 512 + wave * 64;

    v8f acc[4];
#pragma unroll
    for (int j = 0; j < 4; ++j) acc[j] = v8f_zero();

    for (int k0 = 0; k0 < K; k0 += 32) {
        v16bf a = load_a_frag(A, K, m0 + lrow, k0, khalf);
#pragma unroll
        for (int j = 0; j < 4; ++j) {
            v16bf b = load_b_frag(Wt, K, c0 + j * 16 + lrow, k0, khalf);
            acc[j] = wmma_bf16(a, b, acc[j]);
        }
    }

#pragma unroll
    for (int j = 0; j < 4; ++j) {
        const int col = c0 + j * 16 + lrow;
        const float bv = bias[col];
#pragma unroll
        for (int r = 0; r < 8; ++r) {
            const int row = m0 + r + khalf * 8;
            float v = acc[j][r] + bv;
            if (RELU) v = fmaxf(v, 0.0f);
            if (OUT_BF16)
                ((bf16_t*)out)[(size_t)row * N + col] = (bf16_t)v;
            else
                ((float*)out)[(size_t)row * N + col] = v;
        }
    }
}

// ---------------------------------------------------------------------------
// Attention: one workgroup per (batch, 16-query tile).
//   LDS: scores f32 [16][S] (256KB) + Q bf16 [16][D] (16KB) + reductions.
//   Pass 1: waves split key tiles, scores = Q @ K^T (WMMA, K-dim = D).
//   Softmax: cooperative row max / exp / sum in LDS.
//   Pass 2: waves split D columns, O = P @ V (WMMA, K-dim = S), final /l.
// ---------------------------------------------------------------------------
__global__ __launch_bounds__(256)
void attention_kernel(const bf16_t* __restrict__ Qb, const bf16_t* __restrict__ Kb,
                      const bf16_t* __restrict__ Vt, float* __restrict__ attn) {
    extern __shared__ char smem_raw[];
    float*  s_scores = (float*)smem_raw;                                   // [16][S]
    bf16_t* s_q      = (bf16_t*)(smem_raw + (size_t)16 * S_DIM * 4);      // [16][D]
    float*  s_red    = (float*)(smem_raw + (size_t)16 * S_DIM * 4 + 16 * DM * 2); // [16][16]
    float*  s_l      = s_red + 256;                                        // [16]
    float*  s_m      = s_l + 16;                                           // [16]

    const int tid   = threadIdx.x;
    const int wave  = tid >> 5;
    const int lane  = tid & 31;
    const int lrow  = lane & 15;
    const int khalf = lane >> 4;

    const int b  = blockIdx.x / (S_DIM / 16);
    const int qt = blockIdx.x % (S_DIM / 16);
    const bf16_t* Qbase = Qb + (size_t)b * S_DIM * DM + (size_t)qt * 16 * DM;
    const bf16_t* Kbase = Kb + (size_t)b * S_DIM * DM;
    const bf16_t* Vbase = Vt + (size_t)b * DM * S_DIM;
    const float scale = 0.044194173824159216f;   // 1/sqrt(512)

    // Stage 16xD Q tile into LDS (bf16, 16KB) with 16B vector copies.
    {
        const uint4* src = (const uint4*)Qbase;
        uint4*       dst = (uint4*)s_q;
        const int n16 = 16 * DM * 2 / 16;
        for (int i = tid; i < n16; i += 256) dst[i] = src[i];
    }
    __syncthreads();

    // ---- Pass 1: raw scores into LDS --------------------------------------
    for (int kt = wave; kt < S_DIM / 16; kt += 8) {
        v8f acc = v8f_zero();
        const int key = kt * 16 + lrow;
        for (int d0 = 0; d0 < DM; d0 += 32) {
            v16bf a = load_a_frag(s_q, DM, lrow, d0, khalf);
            v16bf bf = load_b_frag(Kbase, DM, key, d0, khalf); // B[k=d][n=key]
            acc = wmma_bf16(a, bf, acc);
        }
#pragma unroll
        for (int r = 0; r < 8; ++r) {
            const int row = r + khalf * 8;
            s_scores[row * S_DIM + kt * 16 + lrow] = acc[r] * scale;
        }
    }
    __syncthreads();

    // ---- Softmax over each of the 16 rows (length S) ----------------------
    {
        const int row = tid >> 4, sub = tid & 15;
        float m = -3.0e38f;
        for (int c = sub; c < S_DIM; c += 16)
            m = fmaxf(m, s_scores[row * S_DIM + c]);
        s_red[row * 16 + sub] = m;
        __syncthreads();
        if (sub == 0) {
            float mm = s_red[row * 16];
#pragma unroll
            for (int i = 1; i < 16; ++i) mm = fmaxf(mm, s_red[row * 16 + i]);
            s_m[row] = mm;
        }
        __syncthreads();
        const float mrow = s_m[row];
        float lsum = 0.0f;
        for (int c = sub; c < S_DIM; c += 16) {
            float e = __expf(s_scores[row * S_DIM + c] - mrow);
            s_scores[row * S_DIM + c] = e;
            lsum += e;
        }
        s_red[row * 16 + sub] = lsum;
        __syncthreads();
        if (sub == 0) {
            float ll = 0.0f;
#pragma unroll
            for (int i = 0; i < 16; ++i) ll += s_red[row * 16 + i];
            s_l[row] = ll;
        }
    }
    __syncthreads();

    // ---- Pass 2: O = P @ V, waves split the 512 output columns ------------
    const int c0 = wave * 64;
    v8f acc[4];
#pragma unroll
    for (int j = 0; j < 4; ++j) acc[j] = v8f_zero();

    for (int kc = 0; kc < S_DIM; kc += 32) {
        v16bf a = load_a_frag(s_scores, S_DIM, lrow, kc, khalf); // f32->bf16
#pragma unroll
        for (int j = 0; j < 4; ++j) {
            v16bf bf = load_b_frag(Vbase, S_DIM, c0 + j * 16 + lrow, kc, khalf);
            acc[j] = wmma_bf16(a, bf, acc[j]);
        }
    }

    float* outb = attn + (size_t)b * S_DIM * DM + (size_t)qt * 16 * DM;
#pragma unroll
    for (int j = 0; j < 4; ++j) {
        const int col = c0 + j * 16 + lrow;
#pragma unroll
        for (int r = 0; r < 8; ++r) {
            const int row = r + khalf * 8;
            outb[(size_t)row * DM + col] = acc[j][r] / s_l[row];
        }
    }
}

// ---------------------------------------------------------------------------
// out = LayerNorm(X + Y) * gamma + beta   — one block per 512-elem row.
// ---------------------------------------------------------------------------
__global__ __launch_bounds__(256)
void add_layernorm_kernel(const float* __restrict__ X, const float* __restrict__ Y,
                          const float* __restrict__ gamma, const float* __restrict__ beta,
                          float* __restrict__ out) {
    __shared__ float red[256];
    const int row = blockIdx.x, tid = threadIdx.x;
    const float* x = X + (size_t)row * DM;
    const float* y = Y + (size_t)row * DM;

    float v0 = x[tid] + y[tid];
    float v1 = x[tid + 256] + y[tid + 256];

    red[tid] = v0 + v1;
    __syncthreads();
    for (int s = 128; s > 0; s >>= 1) {
        if (tid < s) red[tid] += red[tid + s];
        __syncthreads();
    }
    const float mu = red[0] * (1.0f / DM);
    __syncthreads();

    const float d0 = v0 - mu, d1 = v1 - mu;
    red[tid] = d0 * d0 + d1 * d1;
    __syncthreads();
    for (int s = 128; s > 0; s >>= 1) {
        if (tid < s) red[tid] += red[tid + s];
        __syncthreads();
    }
    const float rstd = rsqrtf(red[0] * (1.0f / DM) + LN_EPS);

    out[(size_t)row * DM + tid]       = d0 * rstd * gamma[tid]       + beta[tid];
    out[(size_t)row * DM + tid + 256] = d1 * rstd * gamma[tid + 256] + beta[tid + 256];
}

// ---------------------------------------------------------------------------
extern "C" void kernel_launch(void* const* d_in, const int* in_sizes, int n_in,
                              void* d_out, int out_size, void* d_ws, size_t ws_size,
                              hipStream_t stream) {
    const float* q      = (const float*)d_in[0];
    const float* k      = (const float*)d_in[1];
    const float* v      = (const float*)d_in[2];
    const float* x      = (const float*)d_in[3];
    const float* Wq     = (const float*)d_in[4];
    const float* bq     = (const float*)d_in[5];
    const float* Wk     = (const float*)d_in[6];
    const float* bk     = (const float*)d_in[7];
    const float* Wv     = (const float*)d_in[8];
    const float* bv     = (const float*)d_in[9];
    const float* gamma1 = (const float*)d_in[10];
    const float* beta1  = (const float*)d_in[11];
    const float* W1     = (const float*)d_in[12];
    const float* b1     = (const float*)d_in[13];
    const float* W2     = (const float*)d_in[14];
    const float* b2     = (const float*)d_in[15];
    const float* gamma2 = (const float*)d_in[16];
    const float* beta2  = (const float*)d_in[17];
    float* out = (float*)d_out;

    char* ws = (char*)d_ws;
    size_t off = 0;
    auto take = [&](size_t bytes) -> char* {
        char* p = ws + off;
        off += (bytes + 255) & ~(size_t)255;
        return p;
    };
    bf16_t* Qb  = (bf16_t*)take((size_t)M_ROWS * DM * 2);
    bf16_t* Kb  = (bf16_t*)take((size_t)M_ROWS * DM * 2);
    bf16_t* Vb  = (bf16_t*)take((size_t)M_ROWS * DM * 2);
    bf16_t* Vt  = (bf16_t*)take((size_t)M_ROWS * DM * 2);
    bf16_t* Hid = (bf16_t*)take((size_t)M_ROWS * DF * 2);
    float*  atn = (float*)take((size_t)M_ROWS * DM * 4);
    float*  h   = (float*)take((size_t)M_ROWS * DM * 4);
    float*  f   = (float*)take((size_t)M_ROWS * DM * 4);
    bf16_t* Wqt = (bf16_t*)take((size_t)DM * DM * 2);
    bf16_t* Wkt = (bf16_t*)take((size_t)DM * DM * 2);
    bf16_t* Wvt = (bf16_t*)take((size_t)DM * DM * 2);
    bf16_t* W1t = (bf16_t*)take((size_t)DF * DM * 2);
    bf16_t* W2t = (bf16_t*)take((size_t)DM * DF * 2);

    // ---- weight prep (f32 [K][N] -> bf16 [N][K]) ----
    transpose_w_kernel<<<dim3(DM / 32, DM / 32), 256, 0, stream>>>(Wq, Wqt, DM, DM);
    transpose_w_kernel<<<dim3(DM / 32, DM / 32), 256, 0, stream>>>(Wk, Wkt, DM, DM);
    transpose_w_kernel<<<dim3(DM / 32, DM / 32), 256, 0, stream>>>(Wv, Wvt, DM, DM);
    transpose_w_kernel<<<dim3(DM / 32, DF / 32), 256, 0, stream>>>(W1, W1t, DM, DF);
    transpose_w_kernel<<<dim3(DF / 32, DM / 32), 256, 0, stream>>>(W2, W2t, DF, DM);

    // ---- QKV projections (bf16 outputs for WMMA attention) ----
    gemm_bias_kernel<float, false, true>
        <<<(M_ROWS / 16) * (DM / 512), 256, 0, stream>>>(q, Wqt, bq, Qb, M_ROWS, DM, DM);
    gemm_bias_kernel<float, false, true>
        <<<(M_ROWS / 16) * (DM / 512), 256, 0, stream>>>(k, Wkt, bk, Kb, M_ROWS, DM, DM);
    gemm_bias_kernel<float, false, true>
        <<<(M_ROWS / 16) * (DM / 512), 256, 0, stream>>>(v, Wvt, bv, Vb, M_ROWS, DM, DM);

    // V^T per batch so P@V B-fragments are contiguous in the key dim.
    transpose_v_kernel<<<dim3(S_DIM / 32, DM / 32, B_DIM), 256, 0, stream>>>(Vb, Vt);

    // ---- attention (scores strip lives in 320KB LDS) ----
    const int smem = 16 * S_DIM * 4 + 16 * DM * 2 + (256 + 16 + 16) * 4;
    hipFuncSetAttribute(reinterpret_cast<const void*>(attention_kernel),
                        hipFuncAttributeMaxDynamicSharedMemorySize, smem);
    attention_kernel<<<B_DIM * (S_DIM / 16), 256, smem, stream>>>(Qb, Kb, Vt, atn);

    // ---- add & norm 1 ----
    add_layernorm_kernel<<<M_ROWS, 256, 0, stream>>>(atn, x, gamma1, beta1, h);

    // ---- FFN ----
    gemm_bias_kernel<float, true, true>
        <<<(M_ROWS / 16) * (DF / 512), 256, 0, stream>>>(h, W1t, b1, Hid, M_ROWS, DF, DM);
    gemm_bias_kernel<bf16_t, false, false>
        <<<(M_ROWS / 16) * (DM / 512), 256, 0, stream>>>(Hid, W2t, b2, f, M_ROWS, DM, DF);

    // ---- add & norm 2 ----
    add_layernorm_kernel<<<M_ROWS, 256, 0, stream>>>(h, f, gamma2, beta2, out);
}
